// LSTM_dist_47699906789960
// MI455X (gfx1250) — compile-verified, matrix operand order checked by reading, not run on hardware
//
#include <hip/hip_runtime.h>

#define B_    64
#define T_    4096
#define DIN   3
#define C1_   64
#define H_    128
#define NCLS_ 40
#define NG    (4*H_)      // 512 gate columns (permuted i,f,g,o interleave)
#define KTOT  (C1_+H_)    // 192 = concat K (xt | h)

typedef __bf16 bf16_t;
typedef __attribute__((ext_vector_type(16))) bf16_t v16bf;
typedef __attribute__((ext_vector_type(8)))  float  v8f;

union Frag { v16bf v; uint4 q[2]; };

__device__ __forceinline__ float fsig(float x) { return 1.0f / (1.0f + __expf(-x)); }
__device__ __forceinline__ float ftanh(float x) {
    float e = __expf(-2.0f * fabsf(x));
    float r = (1.0f - e) / (1.0f + e);
    return x < 0.0f ? -r : r;
}

// ---------------------------------------------------------------------------
// 1) Input second-moment reduction: Sx[3], Sxx[6] -> 9 floats (BN is affine in
//    x, so channel stats follow analytically from these 9 scalars).
// ---------------------------------------------------------------------------
__global__ void stats_kernel(const float* __restrict__ x, float* __restrict__ stats, int N)
{
    float s[9];
#pragma unroll
    for (int j = 0; j < 9; ++j) s[j] = 0.0f;
    for (int i = blockIdx.x * blockDim.x + threadIdx.x; i < N; i += gridDim.x * blockDim.x) {
        float x0 = x[3*i + 0], x1 = x[3*i + 1], x2 = x[3*i + 2];
        s[0] += x0;    s[1] += x1;    s[2] += x2;
        s[3] += x0*x0; s[4] += x0*x1; s[5] += x0*x2;
        s[6] += x1*x1; s[7] += x1*x2; s[8] += x2*x2;
    }
#pragma unroll
    for (int off = 16; off > 0; off >>= 1)
#pragma unroll
        for (int j = 0; j < 9; ++j) s[j] += __shfl_down(s[j], off);
    if ((threadIdx.x & 31) == 0)
#pragma unroll
        for (int j = 0; j < 9; ++j) atomicAdd(&stats[j], s[j]);
}

// ---------------------------------------------------------------------------
// 2) BN scale/shift per channel + gate-permuted bf16 weight concat + bias.
//    Permutation: column n -> gate g = n%4 (i,f,g,o), hidden unit j = n/4,
//    original row r = g*128 + j.  Wcat[n][k] = k<64 ? w_ih[r][k] : w_hh[r][k-64]
// ---------------------------------------------------------------------------
__global__ void prep_kernel(const float* __restrict__ conv_w, const float* __restrict__ conv_b,
                            const float* __restrict__ gamma,  const float* __restrict__ beta,
                            const float* __restrict__ w_ih,   const float* __restrict__ b_ih,
                            const float* __restrict__ w_hh,   const float* __restrict__ b_hh,
                            const float* __restrict__ stats,
                            float* __restrict__ scale, float* __restrict__ shift,
                            float* __restrict__ bias_p, bf16_t* __restrict__ Wcat)
{
    const int tid = threadIdx.x;                // 512 threads
    const float invN = 1.0f / (float)(B_ * T_);
    if (tid < C1_) {
        const float* w = conv_w + tid * DIN;
        float b  = conv_b[tid];
        float dm = (w[0]*stats[0] + w[1]*stats[1] + w[2]*stats[2]) * invN;
        float quad = w[0]*w[0]*stats[3] + w[1]*w[1]*stats[6] + w[2]*w[2]*stats[8]
                   + 2.0f*(w[0]*w[1]*stats[4] + w[0]*w[2]*stats[5] + w[1]*w[2]*stats[7]);
        float mean = dm + b;
        float ey2  = quad * invN + 2.0f*b*dm + b*b;
        float var  = ey2 - mean*mean;
        float sc   = gamma[tid] * rsqrtf(var + 1e-5f);
        scale[tid] = sc;
        shift[tid] = beta[tid] - mean * sc;
    }
    const int n = tid;
    const int r = (n & 3) * H_ + (n >> 2);
    bias_p[n] = b_ih[r] + b_hh[r];
    for (int k = 0; k < KTOT; ++k)
        Wcat[n*KTOT + k] = (bf16_t)(k < C1_ ? w_ih[r*C1_ + k] : w_hh[r*H_ + (k - C1_)]);
}

// ---------------------------------------------------------------------------
// 3) Fused conv1x1 + BN + ReLU -> bf16 activations, layout [t][b][c] so each
//    timestep is one contiguous 8 KB tile (stream stays resident in L2).
// ---------------------------------------------------------------------------
__global__ void yact_kernel(const float* __restrict__ x, const float* __restrict__ conv_w,
                            const float* __restrict__ conv_b, const float* __restrict__ scale,
                            const float* __restrict__ shift, bf16_t* __restrict__ yact)
{
    int idx = blockIdx.x * blockDim.x + threadIdx.x;
    if (idx >= T_ * B_ * C1_) return;
    int c = idx & 63;
    int b = (idx >> 6) & 63;
    int t = idx >> 12;
    const float* xp = x + ((size_t)b * T_ + t) * DIN;
    const float* wp = conv_w + c * DIN;
    float y = xp[0]*wp[0] + xp[1]*wp[1] + xp[2]*wp[2] + conv_b[c];
    y = y * scale[c] + shift[c];
    yact[idx] = (bf16_t)fmaxf(y, 0.0f);
}

// ---------------------------------------------------------------------------
// 4) Persistent single-WGP LSTM. 16 waves (4/SIMD for VGPR headroom); wave w
//    owns N-tiles 2w and 2w+1 (16 permuted gate columns each = 4 hidden units
//    with all four gates). B operands re-read from LDS (hoisted over the
//    mt-loop); c lives in LDS; h double-buffered in LDS. 2 barriers/step.
// ---------------------------------------------------------------------------
__global__ __launch_bounds__(512, 1) void lstm_kernel(
    const bf16_t* __restrict__ yact, const bf16_t* __restrict__ Wcat,
    const float* __restrict__ bias_p, const float* __restrict__ h0,
    const float* __restrict__ c0, float* __restrict__ hT)
{
    __shared__ __align__(16) bf16_t Wl[NG * KTOT];    // 192 KB
    __shared__ __align__(16) bf16_t hl[2][B_ * H_];   //  32 KB (double buffer)
    __shared__ __align__(16) bf16_t xts[B_ * C1_];    //   8 KB
    __shared__ __align__(16) float  cl[B_ * H_];      //  32 KB cell state

    const int tid  = threadIdx.x;     // 0..511
    const int w    = tid >> 5;        // wave id 0..15, owns ntiles 2w, 2w+1
    const int l    = tid & 31;        // lane
    const int lmod = l & 15;
    const int lhi  = l >> 4;          // K-half select in A/B fragment layout
    const int koff = lhi * 8;
    const int r4   = lmod & 3;        // gate role within 4-lane group (i,f,g,o)
    const int jloc = lmod >> 2;

    // --- one-time setup -----------------------------------------------------
    for (int i = tid; i < NG * KTOT / 8; i += 512)
        ((uint4*)Wl)[i] = ((const uint4*)Wcat)[i];
    for (int i = tid; i < B_ * H_; i += 512) {
        hl[0][i] = (bf16_t)h0[i];
        cl[i]    = c0[i];
    }
    float bias_h[2];
#pragma unroll
    for (int half = 0; half < 2; ++half)
        bias_h[half] = bias_p[(2*w + half) * 16 + lmod];
    __syncthreads();

    // --- sequential time loop ----------------------------------------------
    for (int t = 0; t < T_; ++t) {
        // stage x_t tile (8 KB) into LDS; prefetch a future step
        ((uint4*)xts)[tid] = ((const uint4*)(yact + (size_t)t * B_ * C1_))[tid];
        if (t + 4 < T_)
            __builtin_prefetch(yact + (size_t)(t + 4) * B_ * C1_ + tid * 8, 0, 1);
        __syncthreads();   // (A) xts ready; prev-step h writes visible

        const bf16_t* hcur = hl[t & 1];
        bf16_t*       hnxt = hl[(t + 1) & 1];

#pragma unroll
        for (int half = 0; half < 2; ++half) {
            const int nt   = 2*w + half;
            const int ncol = nt * 16 + lmod;
            const int j    = nt * 4 + jloc;

            v8f acc[4];
#pragma unroll
            for (int mt = 0; mt < 4; ++mt)
#pragma unroll
                for (int v2 = 0; v2 < 8; ++v2) acc[mt][v2] = bias_h[half];

#pragma unroll
            for (int kt = 0; kt < 6; ++kt) {
                Frag Bt;   // reused across the 4 M-tiles
                const bf16_t* bp = &Wl[ncol * KTOT + kt * 32 + koff];
                Bt.q[0] = *(const uint4*)bp;
                Bt.q[1] = *(const uint4*)(bp + 16);
#pragma unroll
                for (int mt = 0; mt < 4; ++mt) {
                    const int m = mt * 16 + lmod;
                    Frag A;
                    const bf16_t* ap = (kt < 2)
                        ? &xts[m * C1_ + kt * 32 + koff]
                        : &hcur[m * H_ + (kt - 2) * 32 + koff];
                    A.q[0] = *(const uint4*)ap;
                    A.q[1] = *(const uint4*)(ap + 16);
                    acc[mt] = __builtin_amdgcn_wmma_f32_16x16x32_bf16(
                        false, A.v, false, Bt.v, (short)0, acc[mt], false, false);
                }
            }

            // nonlinearity: 4-lane groups hold (i,f,g,o) of one (row, unit)
#pragma unroll
            for (int mt = 0; mt < 4; ++mt) {
#pragma unroll
                for (int v2 = 0; v2 < 8; ++v2) {
                    float a  = acc[mt][v2];
                    float b1 = __shfl_xor(a, 1);
                    float b2 = __shfl_xor(a, 2);
                    float b3 = __shfl_xor(a, 3);
                    float gi, gf, gg, go;
                    if      (r4 == 0) { gi = a;  gf = b1; gg = b2; go = b3; }
                    else if (r4 == 1) { gi = b1; gf = a;  gg = b3; go = b2; }
                    else if (r4 == 2) { gi = b2; gf = b3; gg = a;  go = b1; }
                    else              { gi = b3; gf = b2; gg = b1; go = a;  }
                    if (r4 == 0) {   // owning lane does c-RMW and h write
                        const int m = mt * 16 + v2 + 8 * lhi;
                        float c = cl[m * H_ + j];
                        c = fsig(gf) * c + fsig(gi) * ftanh(gg);
                        cl[m * H_ + j] = c;
                        float h = fsig(go) * ftanh(c);
                        hnxt[m * H_ + j] = (bf16_t)h;
                        if (t == T_ - 1) hT[m * H_ + j] = h;
                    }
                }
            }
        }
        __syncthreads();   // (B) all xts/hcur reads done before next staging
    }
}

// ---------------------------------------------------------------------------
// 5) Tiny classifier: out[64][40] = hT @ out_w.T + out_b
// ---------------------------------------------------------------------------
__global__ void head_kernel(const float* __restrict__ hT, const float* __restrict__ out_w,
                            const float* __restrict__ out_b, float* __restrict__ out)
{
    int idx = blockIdx.x * blockDim.x + threadIdx.x;
    if (idx >= B_ * NCLS_) return;
    int b = idx / NCLS_, c = idx % NCLS_;
    float s = out_b[c];
    for (int k = 0; k < H_; ++k) s += hT[b*H_ + k] * out_w[c*H_ + k];
    out[idx] = s;
}

extern "C" void kernel_launch(void* const* d_in, const int* in_sizes, int n_in,
                              void* d_out, int out_size, void* d_ws, size_t ws_size,
                              hipStream_t stream)
{
    const float* x      = (const float*)d_in[0];
    const float* conv_w = (const float*)d_in[1];
    const float* conv_b = (const float*)d_in[2];
    const float* gamma  = (const float*)d_in[3];
    const float* beta   = (const float*)d_in[4];
    const float* w_ih   = (const float*)d_in[5];
    const float* b_ih   = (const float*)d_in[6];
    const float* w_hh   = (const float*)d_in[7];
    const float* b_hh   = (const float*)d_in[8];
    const float* out_w  = (const float*)d_in[9];
    const float* out_b  = (const float*)d_in[10];
    const float* h0     = (const float*)d_in[11];
    const float* c0     = (const float*)d_in[12];

    char* ws = (char*)d_ws;
    float*  stats  = (float*)(ws + 0);        // 9 f32 (64 B reserved)
    float*  scale  = (float*)(ws + 64);       // 64 f32
    float*  shift  = (float*)(ws + 320);      // 64 f32
    float*  bias_p = (float*)(ws + 576);      // 512 f32
    bf16_t* Wcat   = (bf16_t*)(ws + 4096);    // 512*192 bf16 = 192 KB
    float*  hT     = (float*)(ws + 200704);   // 64*128 f32  = 32 KB
    bf16_t* yact   = (bf16_t*)(ws + 262144);  // 4096*64*64 bf16 = 32 MB

    hipMemsetAsync(stats, 0, 64, stream);
    stats_kernel<<<1024, 256, 0, stream>>>(x, stats, B_ * T_);
    prep_kernel<<<1, 512, 0, stream>>>(conv_w, conv_b, gamma, beta,
                                       w_ih, b_ih, w_hh, b_hh,
                                       stats, scale, shift, bias_p, Wcat);
    yact_kernel<<<(T_ * B_ * C1_) / 256, 256, 0, stream>>>(x, conv_w, conv_b, scale, shift, yact);
    lstm_kernel<<<1, 512, 0, stream>>>(yact, Wcat, bias_p, h0, c0, hT);
    head_kernel<<<(B_ * NCLS_ + 255) / 256, 256, 0, stream>>>(hT, out_w, out_b, (float*)d_out);
}
